// SonataMoEFlow_84593675862654
// MI455X (gfx1250) — matrix operand order repaired and checked by reading, not compiled, not run on hardware
//
#include <hip/hip_runtime.h>
#include <math.h>

typedef __attribute__((ext_vector_type(16))) _Float16 v16h;
typedef __attribute__((ext_vector_type(8)))  _Float16 v8h;
typedef __attribute__((ext_vector_type(8)))  float    v8f;

#define BDIM 1024
#define NTOK 2048   // B*T
#define TT   512
#define NH   16
#define HDIM 64
#define NEXP 8
#define FFD  4096

// ---------------------------------------------------------------------------
// Generic WMMA GEMM:  C[M,N] = act(A[M,K] @ W[K,N] + bias) + res
// Tile 64x128x32, 256 threads (8 waves), wave = 2x2 WMMA 16x16x32 tiles.
// Template: GATHER = A rows via glist (MoE token gather, implies GUARD),
//           GUARD  = runtime M bound (device counts) and K/N edge guards.
// ---------------------------------------------------------------------------
#define BM 64
#define BN 128
#define BK 32
#define LSTR 40   // halves per LDS row: 80B, 16B-aligned chunks, conflict-free

template <bool GATHER, bool GUARD>
__global__ void __launch_bounds__(256)
gemm_kernel(const float* __restrict__ A, const float* __restrict__ W,
            const float* __restrict__ bias, const float* __restrict__ res,
            float* __restrict__ C,
            int M, int Kd, int Ncols, int act,
            const int* __restrict__ glist, const int* __restrict__ gcount,
            long long sA, long long sW, long long sB, long long sC)
{
    __shared__ _Float16 As[BM * LSTR];   // 5120 B
    __shared__ _Float16 Bs[BN * LSTR];   // 10240 B

    const int e = blockIdx.z;
    A += (long long)e * sA;
    W += (long long)e * sW;
    if (bias) bias += (long long)e * sB;
    C += (long long)e * sC;
    const int Mlim = GUARD ? (gcount ? gcount[e] : M) : M;

    const int m0 = blockIdx.y * BM;
    const int n0 = blockIdx.x * BN;
    if (m0 >= Mlim) return;

    const int tid  = threadIdx.x;
    const int lane = tid & 31, l16 = lane & 15, hi = lane >> 4;
    const int wv   = tid >> 5;
    const int wm   = wv >> 2;      // 0..1
    const int wn   = wv & 3;       // 0..3

    // staging ownership
    const int arow = tid >> 2;             // 0..63
    const int ak   = (tid & 3) * 8;        // 0,8,16,24
    const int bk   = tid >> 3;             // 0..31
    const int bn   = (tid & 7) * 16;       // 0..112

    bool arow_ok = true;
    long long arow_g = m0 + arow;
    if (GUARD) arow_ok = (m0 + arow) < Mlim;
    if (GATHER) {
        int idx = glist[e * NTOK + m0 + arow];
        arow_g = arow_ok ? idx : 0;
    }
    const float* Arow = A + arow_g * Kd;

    float4 ra0, ra1, rb0, rb1, rb2, rb3;
    const float4 z4 = make_float4(0.f, 0.f, 0.f, 0.f);

    auto load_tiles = [&](int k0) {
        ra0 = z4; ra1 = z4;
        bool aok = (!GATHER || arow_ok) && (!GUARD || (k0 + ak) < Kd);
        if (aok) {
            const float4* pa = (const float4*)(Arow + k0 + ak);
            ra0 = pa[0]; ra1 = pa[1];
        }
        rb0 = z4; rb1 = z4; rb2 = z4; rb3 = z4;
        bool bok = !GUARD || ((k0 + bk) < Kd && (n0 + bn) < Ncols);
        if (bok) {
            const float4* pb = (const float4*)(W + (long long)(k0 + bk) * Ncols + n0 + bn);
            rb0 = pb[0]; rb1 = pb[1]; rb2 = pb[2]; rb3 = pb[3];
        }
    };
    auto stage_tiles = [&]() {
        v8h pa;
        pa[0] = (_Float16)ra0.x; pa[1] = (_Float16)ra0.y;
        pa[2] = (_Float16)ra0.z; pa[3] = (_Float16)ra0.w;
        pa[4] = (_Float16)ra1.x; pa[5] = (_Float16)ra1.y;
        pa[6] = (_Float16)ra1.z; pa[7] = (_Float16)ra1.w;
        *(v8h*)&As[arow * LSTR + ak] = pa;
        float wbuf[16] = { rb0.x, rb0.y, rb0.z, rb0.w, rb1.x, rb1.y, rb1.z, rb1.w,
                           rb2.x, rb2.y, rb2.z, rb2.w, rb3.x, rb3.y, rb3.z, rb3.w };
#pragma unroll
        for (int j = 0; j < 16; ++j)
            Bs[(bn + j) * LSTR + bk] = (_Float16)wbuf[j];
    };

    v8f acc00 = {}, acc01 = {}, acc10 = {}, acc11 = {};
    const int ksteps = (Kd + BK - 1) / BK;

    load_tiles(0);
    for (int kt = 0; kt < ksteps; ++kt) {
        stage_tiles();
        __syncthreads();
        if (kt + 1 < ksteps) {
            load_tiles((kt + 1) * BK);
            if (kt + 2 < ksteps)
                __builtin_prefetch(W + (long long)((kt + 2) * BK + bk) * Ncols + n0 + bn, 0, 1);
        }
        v16h a0, a1, b0, b1;
        {
            const _Float16* pa0 = &As[(wm * 32 + l16) * LSTR];
            const _Float16* pa1 = &As[(wm * 32 + 16 + l16) * LSTR];
#pragma unroll
            for (int v = 0; v < 8; ++v) {
                int k = ((v >> 2) << 4) + (hi << 3) + ((v & 3) << 1);
                a0[2 * v] = pa0[k]; a0[2 * v + 1] = pa0[k + 1];
                a1[2 * v] = pa1[k]; a1[2 * v + 1] = pa1[k + 1];
            }
            const _Float16* pb0 = &Bs[(wn * 32 + l16) * LSTR];
            const _Float16* pb1 = &Bs[(wn * 32 + 16 + l16) * LSTR];
#pragma unroll
            for (int v = 0; v < 8; ++v) {
                int k = ((v >> 2) << 4) + (hi << 3) + ((v & 3) << 1);
                b0[2 * v] = pb0[k]; b0[2 * v + 1] = pb0[k + 1];
                b1[2 * v] = pb1[k]; b1[2 * v + 1] = pb1[k + 1];
            }
        }
        acc00 = __builtin_amdgcn_wmma_f32_16x16x32_f16(false, a0, false, b0, (short)0, acc00, false, false);
        acc01 = __builtin_amdgcn_wmma_f32_16x16x32_f16(false, a0, false, b1, (short)0, acc01, false, false);
        acc10 = __builtin_amdgcn_wmma_f32_16x16x32_f16(false, a1, false, b0, (short)0, acc10, false, false);
        acc11 = __builtin_amdgcn_wmma_f32_16x16x32_f16(false, a1, false, b1, (short)0, acc11, false, false);
        __syncthreads();
    }

    // epilogue: C layout M = v + hi*8, N = l16
#pragma unroll
    for (int st = 0; st < 4; ++st) {
        const int i = st >> 1, j = st & 1;
        v8f acc = (st == 0) ? acc00 : (st == 1) ? acc01 : (st == 2) ? acc10 : acc11;
#pragma unroll
        for (int v = 0; v < 8; ++v) {
            int gm = m0 + wm * 32 + i * 16 + v + hi * 8;
            int gn = n0 + wn * 32 + j * 16 + l16;
            if (!GUARD || (gm < Mlim && gn < Ncols)) {
                float r = acc[v];
                if (bias) r += bias[gn];
                if (act == 1) r = 0.5f * r * (1.0f + erff(r * 0.70710678118f));
                if (res) r += res[(long long)gm * Ncols + gn];
                C[(long long)gm * Ncols + gn] = r;
            }
        }
    }
}

// ---------------------------------------------------------------------------
// Flash attention over qkv buffer [N, 3*D]. 128 threads = 4 waves per block;
// all waves share (b,h), so K/V tiles are staged cooperatively into LDS once
// per 32-key step (float4 loads + f16 convert), then each wave builds its
// WMMA B-fragments from LDS. Online softmax in 16-lane halves.
// ---------------------------------------------------------------------------
#define KSTR 72   // Ks row stride (halves): 144B rows, conflict-free
#define VSTR 40   // Vs row stride (halves)

__global__ void __launch_bounds__(128)
attn_kernel(const float* __restrict__ qkv, float* __restrict__ o)
{
    __shared__ _Float16 Ks[32 * KSTR];     // K tile  [key][d]      (4608 B)
    __shared__ _Float16 Vs[64 * VSTR];     // V tile  [d][key]      (5120 B)
    __shared__ _Float16 Ps[4][16 * 34];    // per-wave P staging    (4352 B)

    const int b = blockIdx.z, h = blockIdx.y;
    const int tid  = threadIdx.x;
    const int wv   = tid >> 5, lane = tid & 31, l16 = lane & 15, hi = lane >> 4;
    const int q0   = blockIdx.x * 64 + wv * 16;
    const long long rs3 = 3 * BDIM;
    const float* base = qkv + (long long)(b * TT) * rs3 + h * HDIM;

    // staging ownership: key row = tid>>2, d chunk = (tid&3)*16
    const int srow = tid >> 2;
    const int sd   = (tid & 3) * 16;

    // Q fragments (scaled by 1/sqrt(HD) = 0.125)
    v16h aq[2];
#pragma unroll
    for (int ks = 0; ks < 2; ++ks) {
        const float* qrow = base + (long long)(q0 + l16) * rs3;
#pragma unroll
        for (int v = 0; v < 8; ++v) {
            int k = ks * 32 + ((v >> 2) << 4) + (hi << 3) + ((v & 3) << 1);
            aq[ks][2 * v]     = (_Float16)(0.125f * qrow[k]);
            aq[ks][2 * v + 1] = (_Float16)(0.125f * qrow[k + 1]);
        }
    }

    float m[8], lsum[8];
#pragma unroll
    for (int v = 0; v < 8; ++v) { m[v] = -1e30f; lsum[v] = 0.f; }
    v8f o0 = {}, o1 = {}, o2 = {}, o3 = {};

    for (int kt = 0; kt < TT; kt += 32) {
        // ---- cooperative K/V tile staging ----
        {
            const float4* kp4 = (const float4*)(base + (long long)(kt + srow) * rs3 + BDIM + sd);
            float4 f0 = kp4[0], f1 = kp4[1], f2 = kp4[2], f3 = kp4[3];
            v8h h0, h1;
            h0[0] = (_Float16)f0.x; h0[1] = (_Float16)f0.y;
            h0[2] = (_Float16)f0.z; h0[3] = (_Float16)f0.w;
            h0[4] = (_Float16)f1.x; h0[5] = (_Float16)f1.y;
            h0[6] = (_Float16)f1.z; h0[7] = (_Float16)f1.w;
            h1[0] = (_Float16)f2.x; h1[1] = (_Float16)f2.y;
            h1[2] = (_Float16)f2.z; h1[3] = (_Float16)f2.w;
            h1[4] = (_Float16)f3.x; h1[5] = (_Float16)f3.y;
            h1[6] = (_Float16)f3.z; h1[7] = (_Float16)f3.w;
            *(v8h*)&Ks[srow * KSTR + sd]     = h0;
            *(v8h*)&Ks[srow * KSTR + sd + 8] = h1;

            const float4* vp4 = (const float4*)(base + (long long)(kt + srow) * rs3 + 2 * BDIM + sd);
            float4 g0 = vp4[0], g1 = vp4[1], g2 = vp4[2], g3 = vp4[3];
            float vb[16] = { g0.x, g0.y, g0.z, g0.w, g1.x, g1.y, g1.z, g1.w,
                             g2.x, g2.y, g2.z, g2.w, g3.x, g3.y, g3.z, g3.w };
#pragma unroll
            for (int j = 0; j < 16; ++j)
                Vs[(sd + j) * VSTR + srow] = (_Float16)vb[j];
        }
        __syncthreads();

        // ---- S = Q @ K^T (from Ks) ----
        v8f s[2];
#pragma unroll
        for (int sub = 0; sub < 2; ++sub) {
            const _Float16* kp = &Ks[(sub * 16 + l16) * KSTR];
            v16h bk0, bk1;
#pragma unroll
            for (int v = 0; v < 8; ++v) {
                int d = ((v >> 2) << 4) + (hi << 3) + ((v & 3) << 1);
                bk0[2 * v] = kp[d];      bk0[2 * v + 1] = kp[d + 1];
                bk1[2 * v] = kp[32 + d]; bk1[2 * v + 1] = kp[32 + d + 1];
            }
            v8f z = {};
            z = __builtin_amdgcn_wmma_f32_16x16x32_f16(false, aq[0], false, bk0, (short)0, z, false, false);
            z = __builtin_amdgcn_wmma_f32_16x16x32_f16(false, aq[1], false, bk1, (short)0, z, false, false);
            s[sub] = z;
        }
        // ---- online softmax ----
#pragma unroll
        for (int v = 0; v < 8; ++v) {
            float mx = fmaxf(s[0][v], s[1][v]);
            mx = fmaxf(mx, __shfl_xor(mx, 1, 16));
            mx = fmaxf(mx, __shfl_xor(mx, 2, 16));
            mx = fmaxf(mx, __shfl_xor(mx, 4, 16));
            mx = fmaxf(mx, __shfl_xor(mx, 8, 16));
            float mnew = fmaxf(m[v], mx);
            float sc = __expf(m[v] - mnew);
            m[v] = mnew;
            float p0 = __expf(s[0][v] - mnew);
            float p1 = __expf(s[1][v] - mnew);
            s[0][v] = p0; s[1][v] = p1;
            float r = p0 + p1;
            r += __shfl_xor(r, 1, 16);
            r += __shfl_xor(r, 2, 16);
            r += __shfl_xor(r, 4, 16);
            r += __shfl_xor(r, 8, 16);
            lsum[v] = lsum[v] * sc + r;
            o0[v] *= sc; o1[v] *= sc; o2[v] *= sc; o3[v] *= sc;
        }
        // P (C layout) -> LDS (row-major) -> A fragment
        _Float16* ps = Ps[wv];
#pragma unroll
        for (int v = 0; v < 8; ++v) {
            ps[(v + hi * 8) * 34 + l16]      = (_Float16)s[0][v];
            ps[(v + hi * 8) * 34 + 16 + l16] = (_Float16)s[1][v];
        }
        __syncthreads();   // Ps visible; all Ks reads done
        v16h ap;
        {
            const _Float16* pr = &ps[l16 * 34];
#pragma unroll
            for (int v = 0; v < 8; ++v) {
                int k = ((v >> 2) << 4) + (hi << 3) + ((v & 3) << 1);
                ap[2 * v] = pr[k]; ap[2 * v + 1] = pr[k + 1];
            }
        }
        // ---- O += P @ V (from Vs) ----
#pragma unroll
        for (int dt = 0; dt < 4; ++dt) {
            const _Float16* vp = &Vs[(dt * 16 + l16) * VSTR];
            v16h bv;
#pragma unroll
            for (int v = 0; v < 8; ++v) {
                int k = ((v >> 2) << 4) + (hi << 3) + ((v & 3) << 1);
                bv[2 * v] = vp[k]; bv[2 * v + 1] = vp[k + 1];
            }
            v8f* op = (dt == 0) ? &o0 : (dt == 1) ? &o1 : (dt == 2) ? &o2 : &o3;
            *op = __builtin_amdgcn_wmma_f32_16x16x32_f16(false, ap, false, bv, (short)0, *op, false, false);
        }
        __syncthreads();   // all Vs/Ps reads done before next staging
    }

#pragma unroll
    for (int v = 0; v < 8; ++v) {
        float inv = 1.0f / lsum[v];
        int row = q0 + v + hi * 8;
        float* orow = o + (long long)(b * TT + row) * BDIM + h * HDIM;
        orow[0 * 16 + l16] = o0[v] * inv;
        orow[1 * 16 + l16] = o1[v] * inv;
        orow[2 * 16 + l16] = o2[v] * inv;
        orow[3 * 16 + l16] = o3[v] * inv;
    }
}

// ---------------------------------------------------------------------------
// LayerNorm / adaLN.  mode 0: out = ln(x)*(1+ss[:, :D]) + ss[:, D:]
//                     mode 1: out = ln(x)*g + bvec
// ---------------------------------------------------------------------------
__global__ void __launch_bounds__(256)
ln_kernel(const float* __restrict__ x, const float* __restrict__ ss,
          const float* __restrict__ g, const float* __restrict__ bvec,
          float* __restrict__ out, int mode)
{
    const int n = blockIdx.x, tid = threadIdx.x;
    const int lane = tid & 31, wv = tid >> 5;
    const float* xr = x + (long long)n * BDIM;

    float vals[4], s1 = 0.f, s2 = 0.f;
#pragma unroll
    for (int i = 0; i < 4; ++i) {
        float v = xr[tid + i * 256];
        vals[i] = v; s1 += v; s2 += v * v;
    }
    for (int off = 16; off > 0; off >>= 1) {
        s1 += __shfl_xor(s1, off, 32);
        s2 += __shfl_xor(s2, off, 32);
    }
    __shared__ float r1[8], r2[8], stats[2];
    if (lane == 0) { r1[wv] = s1; r2[wv] = s2; }
    __syncthreads();
    if (tid == 0) {
        float a = 0.f, b = 0.f;
        for (int j = 0; j < 8; ++j) { a += r1[j]; b += r2[j]; }
        float mean = a * (1.0f / BDIM);
        float var  = b * (1.0f / BDIM) - mean * mean;
        stats[0] = mean; stats[1] = rsqrtf(var + 1e-5f);
    }
    __syncthreads();
    float mean = stats[0], rinv = stats[1];
#pragma unroll
    for (int i = 0; i < 4; ++i) {
        int c = tid + i * 256;
        float xn = (vals[i] - mean) * rinv;
        float a, sh;
        if (mode == 0) {
            a  = 1.0f + ss[(long long)n * (2 * BDIM) + c];
            sh = ss[(long long)n * (2 * BDIM) + BDIM + c];
        } else {
            a = g[c]; sh = bvec[c];
        }
        out[(long long)n * BDIM + c] = xn * a + sh;
    }
}

// ---------------------------------------------------------------------------
// Router: logits -> softmax -> top-2 -> weights, atomic per-expert lists.
// ---------------------------------------------------------------------------
__global__ void __launch_bounds__(64)
route_kernel(const float* __restrict__ h, const float* __restrict__ rw,
             int* __restrict__ topi, float* __restrict__ topw,
             int* __restrict__ counts, int* __restrict__ lists, int* __restrict__ pos)
{
    const int n = blockIdx.x, tid = threadIdx.x;
    float acc[NEXP];
#pragma unroll
    for (int e = 0; e < NEXP; ++e) acc[e] = 0.f;
    const float* hr = h + (long long)n * BDIM;
    for (int k = tid; k < BDIM; k += 64) {
        float xv = hr[k];
        const float* wr = rw + (long long)k * NEXP;
#pragma unroll
        for (int e = 0; e < NEXP; ++e) acc[e] += xv * wr[e];
    }
    __shared__ float part[64][NEXP];
#pragma unroll
    for (int e = 0; e < NEXP; ++e) part[tid][e] = acc[e];
    __syncthreads();
    if (tid < NEXP) {
        float s = 0.f;
        for (int j = 0; j < 64; ++j) s += part[j][tid];
        part[0][tid] = s;
    }
    __syncthreads();
    if (tid == 0) {
        float lg[NEXP], mx = -1e30f;
        for (int e = 0; e < NEXP; ++e) { lg[e] = part[0][e]; mx = fmaxf(mx, lg[e]); }
        float pr[NEXP], Z = 0.f;
        for (int e = 0; e < NEXP; ++e) { pr[e] = __expf(lg[e] - mx); Z += pr[e]; }
        for (int e = 0; e < NEXP; ++e) pr[e] /= Z;
        int i0 = 0;
        for (int e = 1; e < NEXP; ++e) if (pr[e] > pr[i0]) i0 = e;
        int i1 = -1;
        for (int e = 0; e < NEXP; ++e) if (e != i0 && (i1 < 0 || pr[e] > pr[i1])) i1 = e;
        float wsum = pr[i0] + pr[i1];
        topi[n * 2] = i0; topi[n * 2 + 1] = i1;
        topw[n * 2] = pr[i0] / wsum; topw[n * 2 + 1] = pr[i1] / wsum;
        int p0 = atomicAdd(&counts[i0], 1); lists[i0 * NTOK + p0] = n; pos[n * 2] = p0;
        int p1 = atomicAdd(&counts[i1], 1); lists[i1 * NTOK + p1] = n; pos[n * 2 + 1] = p1;
    }
}

// x[n] += w0*y2[e0][p0] + w1*y2[e1][p1]   (fixed order per token -> deterministic)
__global__ void __launch_bounds__(256)
combine_kernel(float* __restrict__ x, const float* __restrict__ y2,
               const int* __restrict__ topi, const float* __restrict__ topw,
               const int* __restrict__ pos)
{
    const int n = blockIdx.x, tid = threadIdx.x;
    const int e0 = topi[n * 2], e1 = topi[n * 2 + 1];
    const float w0 = topw[n * 2], w1 = topw[n * 2 + 1];
    const long long r0 = ((long long)e0 * NTOK + pos[n * 2]) * BDIM;
    const long long r1 = ((long long)e1 * NTOK + pos[n * 2 + 1]) * BDIM;
    float* xr = x + (long long)n * BDIM;
#pragma unroll
    for (int i = 0; i < 4; ++i) {
        int c = tid + i * 256;
        xr[c] += w0 * y2[r0 + c] + w1 * y2[r1 + c];
    }
}

// time embedding MLP: silu(emb @ w1 + b1) @ w2 + b2   (B=4, CD=512)
__global__ void __launch_bounds__(256)
temb_kernel(const float* __restrict__ t, const float* __restrict__ w1,
            const float* __restrict__ b1, const float* __restrict__ w2,
            const float* __restrict__ b2, float* __restrict__ outv)
{
    __shared__ float emb[512], hid[512];
    const int b = blockIdx.x, tid = threadIdx.x;
    const float tv = t[b];
    {
        float fr  = expf(-9.210340371976184f * (float)tid / 256.0f);
        float ang = tv * fr;
        emb[tid]       = sinf(ang);
        emb[tid + 256] = cosf(ang);
    }
    __syncthreads();
    for (int j = tid; j < 512; j += 256) {
        float a = b1[j];
        for (int k = 0; k < 512; ++k) a += emb[k] * w1[k * 512 + j];
        hid[j] = a / (1.0f + __expf(-a));
    }
    __syncthreads();
    for (int j = tid; j < 512; j += 256) {
        float a = b2[j];
        for (int k = 0; k < 512; ++k) a += hid[k] * w2[k * 512 + j];
        outv[b * 512 + j] = a;
    }
}

// speaker: spk_emb[sid] @ spk_w + spk_b  (256 -> 512)
__global__ void __launch_bounds__(256)
spk_kernel(const int* __restrict__ sid, const float* __restrict__ emb,
           const float* __restrict__ w, const float* __restrict__ bias,
           float* __restrict__ outv)
{
    __shared__ float sv[256];
    const int b = blockIdx.x, tid = threadIdx.x;
    const int s = sid[b];
    sv[tid] = emb[s * 256 + tid];
    __syncthreads();
    for (int j = tid; j < 512; j += 256) {
        float a = bias[j];
        for (int k = 0; k < 256; ++k) a += sv[k] * w[k * 512 + j];
        outv[b * 512 + j] = a;
    }
}

// cond_in[n] = concat(sem_emb[tok[n]], t_emb[b], spk[b])  (1536 floats/row)
// Pure f32 gather-copy via gfx1250 async LDS data movers; per-thread chunk
// ownership means only wave-level s_wait_asynccnt is needed.
__global__ void __launch_bounds__(128)
condin_kernel(const int* __restrict__ tok, const float* __restrict__ sem_emb,
              const float* __restrict__ temb, const float* __restrict__ spk,
              float* __restrict__ out)
{
    extern __shared__ float buf[];   // 1536 floats allocated at launch
    const int n = blockIdx.x, tid = threadIdx.x;
    const int b = n >> 9;
    const int tk = tok[n];
#pragma unroll
    for (int i = 0; i < 3; ++i) {
        int c = tid + i * 128;                    // 16B chunk id
        const float* src;
        if (c < 128)      src = sem_emb + (long long)tk * 512 + c * 4;
        else if (c < 256) src = temb + b * 512 + (c - 128) * 4;
        else              src = spk + b * 512 + (c - 256) * 4;
        unsigned lds = (unsigned)(c * 16);
        asm volatile("global_load_async_to_lds_b128 %0, %1, off"
                     :: "v"(lds), "v"(src) : "memory");
    }
    asm volatile("s_wait_asynccnt 0x0" ::: "memory");
    float* dst = out + (long long)n * 1536;
#pragma unroll
    for (int i = 0; i < 3; ++i) {
        int c = tid + i * 128;
        unsigned lds = (unsigned)(c * 16);
        const float* gdst = dst + c * 4;
        asm volatile("global_store_async_from_lds_b128 %0, %1, off"
                     :: "v"(gdst), "v"(lds) : "memory");
    }
    asm volatile("s_wait_asynccnt 0x0" ::: "memory");
    (void)buf;
}

// ---------------------------------------------------------------------------
extern "C" void kernel_launch(void* const* d_in, const int* in_sizes, int n_in,
                              void* d_out, int out_size, void* d_ws, size_t ws_size,
                              hipStream_t stream)
{
    (void)in_sizes; (void)n_in; (void)out_size; (void)ws_size;

    const float* x_t      = (const float*)d_in[0];
    const float* t        = (const float*)d_in[1];
    const int*   sem_tok  = (const int*)  d_in[2];
    const int*   spk_id   = (const int*)  d_in[3];
    const float* sem_emb  = (const float*)d_in[4];
    const float* t_w1     = (const float*)d_in[5];
    const float* t_b1     = (const float*)d_in[6];
    const float* t_w2     = (const float*)d_in[7];
    const float* t_b2     = (const float*)d_in[8];
    const float* spk_emb  = (const float*)d_in[9];
    const float* spk_w    = (const float*)d_in[10];
    const float* spk_b    = (const float*)d_in[11];
    const float* cond_w   = (const float*)d_in[12];
    const float* cond_b   = (const float*)d_in[13];
    const float* in_w     = (const float*)d_in[14];
    const float* in_b     = (const float*)d_in[15];
    const float* ada1_w   = (const float*)d_in[16];
    const float* ada1_b   = (const float*)d_in[17];
    const float* qkv_w    = (const float*)d_in[18];
    const float* aout_w   = (const float*)d_in[19];
    const float* ada2_w   = (const float*)d_in[20];
    const float* ada2_b   = (const float*)d_in[21];
    const float* router_w = (const float*)d_in[22];
    const float* e_w1     = (const float*)d_in[23];
    const float* e_b1     = (const float*)d_in[24];
    const float* e_w2     = (const float*)d_in[25];
    const float* e_b2     = (const float*)d_in[26];
    const float* ln_g     = (const float*)d_in[27];
    const float* ln_b     = (const float*)d_in[28];
    const float* out_w    = (const float*)d_in[29];
    const float* out_b    = (const float*)d_in[30];

    char* wsb = (char*)d_ws;
    size_t off = 0;
    auto alloc = [&](size_t bytes) -> void* {
        void* p = wsb + off;
        off += (bytes + 255) & ~(size_t)255;
        return p;
    };
    float* cond_in = (float*)alloc((size_t)NTOK * 1536 * 4);
    float* cond    = (float*)alloc((size_t)NTOK * BDIM * 4);
    float* x       = (float*)alloc((size_t)NTOK * BDIM * 4);
    float* xn      = (float*)alloc((size_t)NTOK * BDIM * 4);
    float* ssb     = (float*)alloc((size_t)NTOK * 2 * BDIM * 4);
    float* qkv     = (float*)alloc((size_t)NTOK * 3 * BDIM * 4);
    float* attn_o  = (float*)alloc((size_t)NTOK * BDIM * 4);
    float* temb    = (float*)alloc((size_t)4 * 512 * 4);
    float* spkv    = (float*)alloc((size_t)4 * 512 * 4);
    float* topw    = (float*)alloc((size_t)NTOK * 2 * 4);
    float* h1      = (float*)alloc((size_t)NEXP * NTOK * FFD * 4);
    float* y2      = (float*)alloc((size_t)NEXP * NTOK * BDIM * 4);
    int*   topi    = (int*)alloc((size_t)NTOK * 2 * 4);
    int*   pos     = (int*)alloc((size_t)NTOK * 2 * 4);
    int*   counts  = (int*)alloc((size_t)NEXP * 4);
    int*   lists   = (int*)alloc((size_t)NEXP * NTOK * 4);

    // mode: 0 = fast (no guards), 1 = guarded, 2 = gathered + guarded
    auto gemm = [&](int mode, const float* A, const float* W, const float* bias,
                    const float* res, float* C, int M, int Kd, int N, int act,
                    const int* gl, const int* gc,
                    long long sA, long long sW, long long sB, long long sC, int Z) {
        dim3 g((N + BN - 1) / BN, (M + BM - 1) / BM, Z);
        if (mode == 0)
            gemm_kernel<false, false><<<g, dim3(256), 0, stream>>>(
                A, W, bias, res, C, M, Kd, N, act, gl, gc, sA, sW, sB, sC);
        else if (mode == 1)
            gemm_kernel<false, true><<<g, dim3(256), 0, stream>>>(
                A, W, bias, res, C, M, Kd, N, act, gl, gc, sA, sW, sB, sC);
        else
            gemm_kernel<true, true><<<g, dim3(256), 0, stream>>>(
                A, W, bias, res, C, M, Kd, N, act, gl, gc, sA, sW, sB, sC);
    };

    // conditioning
    temb_kernel<<<dim3(4), dim3(256), 0, stream>>>(t, t_w1, t_b1, t_w2, t_b2, temb);
    spk_kernel<<<dim3(4), dim3(256), 0, stream>>>(spk_id, spk_emb, spk_w, spk_b, spkv);
    condin_kernel<<<dim3(NTOK), dim3(128), 1536 * sizeof(float), stream>>>(
        sem_tok, sem_emb, temb, spkv, cond_in);
    gemm(0, cond_in, cond_w, cond_b, nullptr, cond, NTOK, 1536, BDIM, 0,
         nullptr, nullptr, 0, 0, 0, 0, 1);
    // x = x_t @ in_w + in_b   (K=80 -> guarded)
    gemm(1, x_t, in_w, in_b, nullptr, x, NTOK, 80, BDIM, 0,
         nullptr, nullptr, 0, 0, 0, 0, 1);

    for (int l = 0; l < 2; ++l) {
        const float* a1w = ada1_w + (size_t)l * BDIM * 2 * BDIM;
        const float* a1b = ada1_b + (size_t)l * 2 * BDIM;
        const float* qw  = qkv_w  + (size_t)l * BDIM * 3 * BDIM;
        const float* ow  = aout_w + (size_t)l * BDIM * BDIM;
        const float* a2w = ada2_w + (size_t)l * BDIM * 2 * BDIM;
        const float* a2b = ada2_b + (size_t)l * 2 * BDIM;
        const float* rw  = router_w + (size_t)l * BDIM * NEXP;
        const float* ew1 = e_w1 + (size_t)l * NEXP * BDIM * FFD;
        const float* eb1 = e_b1 + (size_t)l * NEXP * FFD;
        const float* ew2 = e_w2 + (size_t)l * NEXP * FFD * BDIM;
        const float* eb2 = e_b2 + (size_t)l * NEXP * BDIM;

        // --- attention block ---
        gemm(0, cond, a1w, a1b, nullptr, ssb, NTOK, BDIM, 2 * BDIM, 0,
             nullptr, nullptr, 0, 0, 0, 0, 1);
        ln_kernel<<<dim3(NTOK), dim3(256), 0, stream>>>(x, ssb, nullptr, nullptr, xn, 0);
        gemm(0, xn, qw, nullptr, nullptr, qkv, NTOK, BDIM, 3 * BDIM, 0,
             nullptr, nullptr, 0, 0, 0, 0, 1);
        attn_kernel<<<dim3(TT / 64, NH, 4), dim3(128), 0, stream>>>(qkv, attn_o);
        gemm(0, attn_o, ow, nullptr, x, x, NTOK, BDIM, BDIM, 0,
             nullptr, nullptr, 0, 0, 0, 0, 1);

        // --- MoE block ---
        gemm(0, cond, a2w, a2b, nullptr, ssb, NTOK, BDIM, 2 * BDIM, 0,
             nullptr, nullptr, 0, 0, 0, 0, 1);
        ln_kernel<<<dim3(NTOK), dim3(256), 0, stream>>>(x, ssb, nullptr, nullptr, xn, 0);
        hipMemsetAsync(counts, 0, NEXP * sizeof(int), stream);
        route_kernel<<<dim3(NTOK), dim3(64), 0, stream>>>(xn, rw, topi, topw,
                                                          counts, lists, pos);
        gemm(2, xn, ew1, eb1, nullptr, h1, NTOK, BDIM, FFD, 1,
             lists, counts,
             0, (long long)BDIM * FFD, FFD, (long long)NTOK * FFD, NEXP);
        gemm(1, h1, ew2, eb2, nullptr, y2, NTOK, FFD, BDIM, 0,
             nullptr, counts,
             (long long)NTOK * FFD, (long long)FFD * BDIM, BDIM,
             (long long)NTOK * BDIM, NEXP);
        combine_kernel<<<dim3(NTOK), dim3(256), 0, stream>>>(x, y2, topi, topw, pos);
    }

    // final LN + output projection (N=80 -> guarded)
    ln_kernel<<<dim3(NTOK), dim3(256), 0, stream>>>(x, nullptr, ln_g, ln_b, xn, 1);
    gemm(1, xn, out_w, out_b, nullptr, (float*)d_out, NTOK, BDIM, 80, 0,
         nullptr, nullptr, 0, 0, 0, 0, 1);
}
